// S4D_60705067762284
// MI455X (gfx1250) — matrix operand order repaired and checked by reading, not compiled, not run on hardware
//
#include <hip/hip_runtime.h>
#include <hip/hip_bf16.h>
#include <math.h>

typedef __attribute__((ext_vector_type(16))) __bf16 v16bf;
typedef __attribute__((ext_vector_type(8)))  __bf16 v8bf;
typedef __attribute__((ext_vector_type(8)))  float  v8f;

constexpr int Bn = 8, Hn = 512, Ln = 4096, NM = 32;  // NM = N/2 complex modes
constexpr int OH = 2 * Hn;
constexpr int CT = 64;             // chunk length (timesteps)
constexpr int NC = Ln / CT;        // 64 chunks per row
constexpr int COLS = Bn * NC;      // 512 GEMM columns per h
constexpr int SV = 2 * NM;         // 64 = real state vector length
constexpr int KC = CT + SV;        // 128 = K of fused chunk GEMM

// ---- workspace layout (bytes) ----
constexpr size_t SZ_YG  = (size_t)Bn * Hn * Ln * 4;        // ygelu f32
constexpr size_t SZ_SS  = (size_t)Hn * COLS * SV * 4;      // state-inc / states f32
constexpr size_t SZ_P   = (size_t)Hn * NM * 16;            // float4 per mode
constexpr size_t SZ_KV  = (size_t)Hn * CT * 4;             // kernel prefix f32
constexpr size_t SZ_AT  = (size_t)Hn * CT * KC * 2;        // [T|V] bf16
constexpr size_t SZ_E   = (size_t)Hn * SV * CT * 2;        // E bf16
constexpr size_t SZ_W   = (size_t)OH * Hn * 2;             // W bf16
constexpr size_t OFF_YG  = 0;
constexpr size_t OFF_SS  = OFF_YG + SZ_YG;
constexpr size_t OFF_P0  = OFF_SS + SZ_SS;
constexpr size_t OFF_P1  = OFF_P0 + SZ_P;
constexpr size_t OFF_KV  = OFF_P1 + SZ_P;
constexpr size_t OFF_ATH = OFF_KV + SZ_KV;
constexpr size_t OFF_ATL = OFF_ATH + SZ_AT;
constexpr size_t OFF_EH  = OFF_ATL + SZ_AT;
constexpr size_t OFF_EL  = OFF_EH + SZ_E;
constexpr size_t OFF_WH  = OFF_EL + SZ_E;
constexpr size_t OFF_WL  = OFF_WH + SZ_W;

// ---------------------------------------------------------------------------
// helpers
// ---------------------------------------------------------------------------
__device__ __forceinline__ v16bf afrag(const __bf16* rowbase, int off) {
  // A fragment K-layout: elements 0-15 at K=k0+kh*16+e (off = k0+kh*8)
  v8bf a = *(const v8bf*)(rowbase + off + (off & 8));
  v8bf b = *(const v8bf*)(rowbase + off + (off & 8) + 8);
  return __builtin_shufflevector(a, b, 0,1,2,3,4,5,6,7,8,9,10,11,12,13,14,15);
}

__device__ __forceinline__ v8f wmma_split(v16bf ah, v16bf al, v16bf bh, v16bf bl, v8f acc) {
  acc = __builtin_amdgcn_wmma_f32_16x16x32_bf16(false, ah, false, bh, (short)0, acc, false, false);
  acc = __builtin_amdgcn_wmma_f32_16x16x32_bf16(false, al, false, bh, (short)0, acc, false, false);
  acc = __builtin_amdgcn_wmma_f32_16x16x32_bf16(false, ah, false, bl, (short)0, acc, false, false);
  return acc;
}

__device__ __forceinline__ void bsplit16(const float* p, v16bf& bh, v16bf& bl) {
  // 16 contiguous f32 -> bf16 hi/lo split (B fragment: lane K = kh*16 + e)
#pragma unroll
  for (int e = 0; e < 16; e += 4) {
    float4 f = *(const float4*)(p + e);
    float vs[4] = {f.x, f.y, f.z, f.w};
#pragma unroll
    for (int j = 0; j < 4; ++j) {
      __bf16 hv = (__bf16)vs[j];
      bh[e + j] = hv;
      bl[e + j] = (__bf16)(vs[j] - (float)hv);
    }
  }
}

// ---------------------------------------------------------------------------
// K1: per-(h,n) mode constants: P0={zr,zi,wr,wi} (w includes the 2x), P1={dtr,dti}
// ---------------------------------------------------------------------------
__global__ void s4d_prep_modes(const float* __restrict__ C, const float* __restrict__ log_dt,
                               const float* __restrict__ lar, const float* __restrict__ aim,
                               float4* __restrict__ P0, float4* __restrict__ P1) {
  int i = blockIdx.x * 256 + threadIdx.x;            // < Hn*NM
  int h = i >> 5;
  float dt  = expf(log_dt[h]);
  float ar  = -expf(lar[i]);
  float ai  = aim[i];
  float cr  = C[2*i], ci = C[2*i+1];
  float dtr = ar * dt, dti = ai * dt;
  float er  = expf(dtr);
  float zr  = er * cosf(dti), zi = er * sinf(dti);
  float inv  = 1.0f / (ar*ar + ai*ai);
  float em1r = zr - 1.0f, em1i = zi;
  float qr = (em1r*ar + em1i*ai) * inv;
  float qi = (em1i*ar - em1r*ai) * inv;
  P0[i] = make_float4(zr, zi, 2.0f*(cr*qr - ci*qi), 2.0f*(cr*qi + ci*qr));
  P1[i] = make_float4(dtr, dti, 0.f, 0.f);
}

// K2: split W_out to bf16 hi/lo
__global__ void s4d_wsplit(const float* __restrict__ W, __bf16* __restrict__ whi,
                           __bf16* __restrict__ wlo) {
  int i = blockIdx.x * 256 + threadIdx.x;            // < OH*Hn
  float w = W[i];
  __bf16 hv = (__bf16)w;
  whi[i] = hv; wlo[i] = (__bf16)(w - (float)hv);
}

// K3: kernel prefix kvec[h][t] = sum_n Re(w_n z_n^t), t<CT
__global__ void s4d_gen_kvec(const float4* __restrict__ P0, const float4* __restrict__ P1,
                             float* __restrict__ kvec) {
  int i = blockIdx.x * 256 + threadIdx.x;            // < Hn*CT
  int h = i >> 6; float t = (float)(i & 63);
  float acc = 0.f;
  for (int n = 0; n < NM; ++n) {
    float4 p0 = P0[h*NM + n], p1 = P1[h*NM + n];
    float e = expf(p1.x * t);
    float cr = e * cosf(p1.y * t), ci = e * sinf(p1.y * t);
    acc += p0.z * cr - p0.w * ci;
  }
  kvec[i] = acc;
}

// K4: AT[h] = [Toeplitz(kvec) | V],  64 x 128, bf16 hi/lo
__global__ void s4d_gen_AT(const float4* __restrict__ P0, const float4* __restrict__ P1,
                           const float* __restrict__ kvec,
                           __bf16* __restrict__ ATh, __bf16* __restrict__ ATl) {
  int i = blockIdx.x * 256 + threadIdx.x;            // < Hn*CT*KC, [h][r][c]
  int c = i & (KC - 1);
  int r = (i >> 7) & (CT - 1);
  int h = i >> 13;
  float val;
  if (c < CT) {
    val = (r >= c) ? kvec[h*CT + (r - c)] : 0.0f;
  } else {
    int q = c - CT; int n = q >> 1;
    float4 p0 = P0[h*NM + n], p1 = P1[h*NM + n];
    float p = (float)(r + 1);
    float e = expf(p1.x * p);
    float zr = e * cosf(p1.y * p), zi = e * sinf(p1.y * p);
    float Qr = p0.z*zr - p0.w*zi;
    float Qi = p0.z*zi + p0.w*zr;
    val = (q & 1) ? -Qi : Qr;
  }
  __bf16 hv = (__bf16)val;
  ATh[i] = hv; ATl[i] = (__bf16)(val - (float)hv);
}

// K5: E[h] 64x64: row 2n = Re(z^{63-t}), row 2n+1 = Im(z^{63-t})
__global__ void s4d_gen_E(const float4* __restrict__ P1, __bf16* __restrict__ Eh,
                          __bf16* __restrict__ El) {
  int i = blockIdx.x * 256 + threadIdx.x;            // < Hn*SV*CT, [h][m][t]
  int t = i & 63; int m = (i >> 6) & 63; int h = i >> 12;
  float4 p1 = P1[h*NM + (m >> 1)];
  float p = (float)(63 - t);
  float e = expf(p1.x * p);
  float zr = e * cosf(p1.y * p), zi = e * sinf(p1.y * p);
  float val = (m & 1) ? zi : zr;
  __bf16 hv = (__bf16)val;
  Eh[i] = hv; El[i] = (__bf16)(val - (float)hv);
}

// ---------------------------------------------------------------------------
// K6: state increments per chunk:  SS[h][col][0..63] = E_h(64x64) @ u_chunks(64x512)
// col = b*64 + chunk; one wave = 64(M) x 16(N) tile, K=64, split-bf16 WMMA
// ---------------------------------------------------------------------------
__global__ void __launch_bounds__(256)
s4d_gemm_E(const float* __restrict__ u, const __bf16* __restrict__ Eh,
           const __bf16* __restrict__ El, float* __restrict__ SS) {
  const int lane = threadIdx.x & 31;
  const int gid  = blockIdx.x * 8 + (threadIdx.x >> 5);
  const int nt = gid & 31, h = gid >> 5;
  const int lc = lane & 15, kh = lane >> 4;
  const int col = nt * 16 + lc;
  const int b = col >> 6, c = col & 63;
  const float* ub = u + (size_t)(b*Hn + h) * Ln + c*CT + kh*16;
  v8f zero = {};
  v8f acc[4] = {zero, zero, zero, zero};
  for (int kt = 0; kt < 2; ++kt) {
    v16bf bh, bl;
    bsplit16(ub + kt*32, bh, bl);
#pragma unroll
    for (int mt = 0; mt < 4; ++mt) {
      const __bf16* rh = Eh + ((size_t)h*64 + mt*16 + lc) * 64;
      const __bf16* rl = El + ((size_t)h*64 + mt*16 + lc) * 64;
      v16bf ah = afrag(rh, kt*32 + kh*8);
      v16bf al = afrag(rl, kt*32 + kh*8);
      acc[mt] = wmma_split(ah, al, bh, bl, acc[mt]);
    }
  }
  float* outp = SS + ((size_t)h*COLS + col) * SV;
#pragma unroll
  for (int mt = 0; mt < 4; ++mt)
#pragma unroll
    for (int r = 0; r < 8; ++r)
      outp[mt*16 + r + 8*kh] = acc[mt][r];
}

// ---------------------------------------------------------------------------
// K7: chunk-level scan (in place): SS[col] becomes state BEFORE chunk.
// One wave per (b,h); lane = mode; only 64 serial steps.
// ---------------------------------------------------------------------------
__global__ void __launch_bounds__(256)
s4d_chunk_scan(const float4* __restrict__ P0, float* __restrict__ SS) {
  const int wave = threadIdx.x >> 5, lane = threadIdx.x & 31;
  const int pair = blockIdx.x * 8 + wave;
  const int b = pair >> 9, h = pair & (Hn - 1);
  float4 p0 = P0[h*NM + lane];
  float ar = p0.x, ai = p0.y;                        // z -> z^64 by 6 squarings
#pragma unroll
  for (int q = 0; q < 6; ++q) { float nr = ar*ar - ai*ai, ni = 2.f*ar*ai; ar = nr; ai = ni; }
  float Sr = 0.f, Si = 0.f;
  float* base = SS + ((size_t)h*COLS + b*NC) * SV + 2*lane;
  for (int c = 0; c < NC; ++c) {
    float2* p = (float2*)(base + (size_t)c * SV);
    float2 inc = *p;
    *p = make_float2(Sr, Si);
    float nSr = ar*Sr - ai*Si + inc.x;
    float nSi = ar*Si + ai*Sr + inc.y;
    Sr = nSr; Si = nSi;
  }
}

// ---------------------------------------------------------------------------
// K8: fused chunk GEMM: y = [T|V] @ [u ; S]  (+ D*u skip, erf-GELU) -> ygelu
// ---------------------------------------------------------------------------
__global__ void __launch_bounds__(256)
s4d_gemm_chunk(const float* __restrict__ u, const float* __restrict__ SS,
               const __bf16* __restrict__ ATh, const __bf16* __restrict__ ATl,
               const float* __restrict__ Dv, float* __restrict__ yg) {
  const int lane = threadIdx.x & 31;
  const int gid  = blockIdx.x * 8 + (threadIdx.x >> 5);
  const int nt = gid & 31, h = gid >> 5;
  const int lc = lane & 15, kh = lane >> 4;
  const int col = nt * 16 + lc;
  const int b = col >> 6, c = col & 63;
  const float* ubase = u + (size_t)(b*Hn + h) * Ln + c*CT;
  const float* ub = ubase + kh*16;
  const float* sb = SS + ((size_t)h*COLS + col) * SV + kh*16;
  v8f zero = {};
  v8f acc[4] = {zero, zero, zero, zero};
  for (int kt = 0; kt < 4; ++kt) {
    v16bf bh, bl;
    const float* src = (kt < 2) ? (ub + kt*32) : (sb + (kt - 2)*32);
    bsplit16(src, bh, bl);
#pragma unroll
    for (int mt = 0; mt < 4; ++mt) {
      const __bf16* rh = ATh + ((size_t)h*CT + mt*16 + lc) * KC;
      const __bf16* rl = ATl + ((size_t)h*CT + mt*16 + lc) * KC;
      v16bf ah = afrag(rh, kt*32 + kh*8);
      v16bf al = afrag(rl, kt*32 + kh*8);
      acc[mt] = wmma_split(ah, al, bh, bl, acc[mt]);
    }
  }
  const float dh = Dv[h];
  float* yp = yg + (size_t)(b*Hn + h) * Ln + c*CT;
#pragma unroll
  for (int mt = 0; mt < 4; ++mt) {
    int mb = mt*16 + 8*kh;
    float4 u0 = *(const float4*)(ubase + mb);
    float4 u1 = *(const float4*)(ubase + mb + 4);
    float uv[8] = {u0.x, u0.y, u0.z, u0.w, u1.x, u1.y, u1.z, u1.w};
    float gv[8];
#pragma unroll
    for (int r = 0; r < 8; ++r) {
      float yv = acc[mt][r] + dh * uv[r];
      gv[r] = 0.5f * yv * (1.0f + erff(yv * 0.70710678118654752f));
    }
    *(float4*)(yp + mb)     = make_float4(gv[0], gv[1], gv[2], gv[3]);
    *(float4*)(yp + mb + 4) = make_float4(gv[4], gv[5], gv[6], gv[7]);
  }
}

// ---------------------------------------------------------------------------
// K9: z = W_out @ ygelu + bias, fused GLU (rows o and o+H share the B fragment)
// ---------------------------------------------------------------------------
__global__ void __launch_bounds__(256)
s4d_gemm_glu(const float* __restrict__ yg, const __bf16* __restrict__ whi,
             const __bf16* __restrict__ wlo, const float* __restrict__ bias,
             float* __restrict__ out) {
  const int lane = threadIdx.x & 31;
  const int gid  = blockIdx.x * 8 + (threadIdx.x >> 5);
  const int nt = gid & 255;
  const int mt = (gid >> 8) & 31;
  const int b  = gid >> 13;
  const int n0 = nt * 16, m0 = mt * 16;
  const int lc = lane & 15, kh = lane >> 4;

  v8f accA = {}, accB = {};
  const float*  ybase = yg + (size_t)b * Hn * Ln + (size_t)(kh*16) * Ln + (n0 + lc);
  const __bf16* wah = whi + (size_t)(m0 + lc) * Hn;
  const __bf16* wal = wlo + (size_t)(m0 + lc) * Hn;
  const __bf16* wbh = whi + (size_t)(m0 + lc + Hn) * Hn;
  const __bf16* wbl = wlo + (size_t)(m0 + lc + Hn) * Hn;

  for (int kt = 0; kt < Hn / 32; ++kt) {
    const int k0 = kt * 32;
    v16bf bh, bl;
    const float* yb = ybase + (size_t)k0 * Ln;
#pragma unroll
    for (int e = 0; e < 16; ++e) {
      float v = yb[(size_t)e * Ln];
      __bf16 hv = (__bf16)v;
      bh[e] = hv;
      bl[e] = (__bf16)(v - (float)hv);
    }
    if (kt + 1 < Hn / 32)
      __builtin_prefetch(ybase + (size_t)(k0 + 32) * Ln, 0, 1);
    v16bf ahA = afrag(wah + k0, kh*8), alA = afrag(wal + k0, kh*8);
    v16bf ahB = afrag(wbh + k0, kh*8), alB = afrag(wbl + k0, kh*8);
    accA = wmma_split(ahA, alA, bh, bl, accA);
    accB = wmma_split(ahB, alB, bh, bl, accB);
  }
#pragma unroll
  for (int r = 0; r < 8; ++r) {
    int o = m0 + r + 8*kh;
    float za = accA[r] + bias[o];
    float zb = accB[r] + bias[o + Hn];
    float sg = 1.0f / (1.0f + expf(-zb));
    out[(size_t)(b*Hn + o) * Ln + (n0 + lc)] = za * sg;
  }
}

// ---------------------------------------------------------------------------
extern "C" void kernel_launch(void* const* d_in, const int* in_sizes, int n_in,
                              void* d_out, int out_size, void* d_ws, size_t ws_size,
                              hipStream_t stream) {
  (void)in_sizes; (void)n_in; (void)out_size; (void)ws_size;
  const float* u   = (const float*)d_in[0];
  const float* C   = (const float*)d_in[1];
  const float* ldt = (const float*)d_in[2];
  const float* lar = (const float*)d_in[3];
  const float* aim = (const float*)d_in[4];
  const float* Dv  = (const float*)d_in[5];
  const float* Wo  = (const float*)d_in[6];
  const float* bo  = (const float*)d_in[7];
  float* out = (float*)d_out;

  char* ws = (char*)d_ws;
  float*   yg  = (float*)(ws + OFF_YG);
  float*   SS  = (float*)(ws + OFF_SS);
  float4*  P0  = (float4*)(ws + OFF_P0);
  float4*  P1  = (float4*)(ws + OFF_P1);
  float*   kv  = (float*)(ws + OFF_KV);
  __bf16*  ATh = (__bf16*)(ws + OFF_ATH);
  __bf16*  ATl = (__bf16*)(ws + OFF_ATL);
  __bf16*  Eh  = (__bf16*)(ws + OFF_EH);
  __bf16*  El  = (__bf16*)(ws + OFF_EL);
  __bf16*  Wh  = (__bf16*)(ws + OFF_WH);
  __bf16*  Wl  = (__bf16*)(ws + OFF_WL);

  s4d_prep_modes<<<(Hn*NM)/256, 256, 0, stream>>>(C, ldt, lar, aim, P0, P1);
  s4d_wsplit    <<<(OH*Hn)/256, 256, 0, stream>>>(Wo, Wh, Wl);
  s4d_gen_kvec  <<<(Hn*CT)/256, 256, 0, stream>>>(P0, P1, kv);
  s4d_gen_AT    <<<(Hn*CT*KC)/256, 256, 0, stream>>>(P0, P1, kv, ATh, ATl);
  s4d_gen_E     <<<(Hn*SV*CT)/256, 256, 0, stream>>>(P1, Eh, El);
  s4d_gemm_E    <<<(Hn*32)/8, 256, 0, stream>>>(u, Eh, El, SS);
  s4d_chunk_scan<<<(Bn*Hn)/8, 256, 0, stream>>>(P0, SS);
  s4d_gemm_chunk<<<(Hn*32)/8, 256, 0, stream>>>(u, SS, ATh, ATl, Dv, yg);
  s4d_gemm_glu  <<<(Bn*32*256)/8, 256, 0, stream>>>(yg, Wh, Wl, bo, out);
}
